// VectorizedMoE_75453985456666
// MI455X (gfx1250) — compile-verified, matrix operand order checked by reading, not run on hardware
//
#include <hip/hip_runtime.h>
#include <hip/hip_bf16.h>

// Problem constants (reference: B=8,S=4096,H=1024,E=8,I=2048,K=2,CAP_FACTOR=1)
#define Hdim   1024
#define Edim   8
#define Idim   2048
#define Ntok   32768            // B*S
#define CAP    4096             // ceil(Ntok/E * 1.0)
#define TM     32               // token tile per block (two 16-row WMMA M tiles)
#define ICHUNK 256              // intermediate chunk kept in LDS between the GEMMs
#define NTHR   512              // 16 waves

typedef __bf16 bf16;
typedef __attribute__((ext_vector_type(16))) __bf16 v16bf;
typedef __attribute__((ext_vector_type(8)))  __bf16 v8bf;
typedef __attribute__((ext_vector_type(8)))  float  v8f;

// ---------------------------------------------------------------------------
// Fragment loaders for V_WMMA_F32_16X16X32_BF16 (wave32).
// A (16x32, 16-bit): lanes 0-15 hold row=lane, K={0..7,16..23};
//                    lanes 16-31 hold row=lane-16, K={8..15,24..31}.
// B (32x16) mirrors A over B^T rows (lane -> N row of W), same K split.
// rowp = row base pointer, k0 = kstep + (lane>>4)*8.
// ---------------------------------------------------------------------------
__device__ inline v16bf frag_bf16(const bf16* rowp, int k0) {
  union { v16bf v; v8bf h[2]; } u;
  u.h[0] = *(const v8bf*)(rowp + k0);        // 16B-aligned b128 load (ds or global)
  u.h[1] = *(const v8bf*)(rowp + k0 + 16);
  return u.v;
}

__device__ inline v16bf frag_gf32(const float* rowp, int k0) {
  const float4 f0 = *(const float4*)(rowp + k0);
  const float4 f1 = *(const float4*)(rowp + k0 + 4);
  const float4 f2 = *(const float4*)(rowp + k0 + 16);
  const float4 f3 = *(const float4*)(rowp + k0 + 20);
  v16bf r;
  r[0]=(bf16)f0.x;  r[1]=(bf16)f0.y;  r[2]=(bf16)f0.z;  r[3]=(bf16)f0.w;
  r[4]=(bf16)f1.x;  r[5]=(bf16)f1.y;  r[6]=(bf16)f1.z;  r[7]=(bf16)f1.w;
  r[8]=(bf16)f2.x;  r[9]=(bf16)f2.y;  r[10]=(bf16)f2.z; r[11]=(bf16)f2.w;
  r[12]=(bf16)f3.x; r[13]=(bf16)f3.y; r[14]=(bf16)f3.z; r[15]=(bf16)f3.w;
  return r;
}

// ---------------------------------------------------------------------------
// Kernel 0: zero the output (scatter-add target; harness poisons d_out).
// ---------------------------------------------------------------------------
__global__ void moe_zero_kernel(float* __restrict__ out, long n4) {
  long i = (long)blockIdx.x * blockDim.x + threadIdx.x;
  long stride = (long)gridDim.x * blockDim.x;
  float4 z = make_float4(0.f, 0.f, 0.f, 0.f);
  for (; i < n4; i += stride) ((float4*)out)[i] = z;
}

// ---------------------------------------------------------------------------
// Kernel 0b: one-time f32 -> bf16 weight cast into workspace.
// ---------------------------------------------------------------------------
__global__ void moe_wcast_kernel(const float* __restrict__ src,
                                 bf16* __restrict__ dst, long n) {
  long i = ((long)blockIdx.x * blockDim.x + threadIdx.x) * 4;
  long stride = (long)gridDim.x * blockDim.x * 4;
  for (; i < n; i += stride) {
    float4 f = *(const float4*)(src + i);
    dst[i + 0] = (bf16)f.x;
    dst[i + 1] = (bf16)f.y;
    dst[i + 2] = (bf16)f.z;
    dst[i + 3] = (bf16)f.w;
  }
}

// ---------------------------------------------------------------------------
// Kernel 1: router. One wave32 per token, embeddings staged in LDS (32 KB).
// ---------------------------------------------------------------------------
__global__ __launch_bounds__(256) void moe_router_kernel(
    const float* __restrict__ x, const float* __restrict__ emb,
    float* __restrict__ weights)
{
  __shared__ float sEmb[Edim * Hdim];
  const int tid = threadIdx.x;
  for (int v = tid; v < Edim * Hdim; v += 256) sEmb[v] = emb[v];
  __syncthreads();

  const int lane = tid & 31, wave = tid >> 5;
  const int token = blockIdx.x * 8 + wave;
  const float* xr = x + (size_t)token * Hdim;

  float acc[Edim];
#pragma unroll
  for (int e = 0; e < Edim; ++e) acc[e] = 0.f;
  for (int h = lane; h < Hdim; h += 32) {
    float xv = xr[h];
#pragma unroll
    for (int e = 0; e < Edim; ++e) acc[e] += xv * sEmb[e * Hdim + h];
  }
#pragma unroll
  for (int e = 0; e < Edim; ++e) {
#pragma unroll
    for (int off = 16; off > 0; off >>= 1)
      acc[e] += __shfl_xor(acc[e], off, 32);
  }
  if (lane == 0) {
    int e1 = 0; float v1 = acc[0];
    for (int e = 1; e < Edim; ++e) if (acc[e] > v1) { v1 = acc[e]; e1 = e; }
    int e2 = -1; float v2 = -3.4e38f;
    for (int e = 0; e < Edim; ++e) if (e != e1 && acc[e] > v2) { v2 = acc[e]; e2 = e; }
    float ex  = __expf(v2 - v1);      // stable 2-way softmax (v1 >= v2)
    float inv = 1.f / (1.f + ex);
    float w1 = inv, w2 = ex * inv;
#pragma unroll
    for (int e = 0; e < Edim; ++e)
      weights[(size_t)token * Edim + e] = (e == e1) ? w1 : ((e == e2) ? w2 : 0.f);
  }
}

// ---------------------------------------------------------------------------
// Kernel 2: capacity assignment. One block per expert; sequential chunk scan
// (wave ballot + LDS prefix) == "first CAP routed tokens in token order".
// ---------------------------------------------------------------------------
__global__ __launch_bounds__(1024) void moe_assign_kernel(
    const float* __restrict__ weights, int* __restrict__ idxBuf,
    float* __restrict__ valBuf)
{
  __shared__ int wsum[32];
  __shared__ int wpre[32];
  __shared__ int sBase;
  __shared__ int sTot;
  const int e = blockIdx.x;
  const int tid = threadIdx.x;
  const int lane = tid & 31, wv = tid >> 5;
  if (tid == 0) sBase = 0;
  __syncthreads();

  for (int chunk = 0; chunk < Ntok / 1024; ++chunk) {
    int token = chunk * 1024 + tid;
    float w = weights[(size_t)token * Edim + e];
    bool f = w > 0.f;
    unsigned bal = (unsigned)__ballot(f);           // wave32: low 32 bits
    int lanePre = __popc(bal & ((1u << lane) - 1u));
    if (lane == 0) wsum[wv] = __popc(bal);
    __syncthreads();
    if (tid == 0) {
      int r = 0;
      for (int i = 0; i < 32; ++i) { wpre[i] = r; r += wsum[i]; }
      sTot = r;
    }
    __syncthreads();
    int slot = sBase + wpre[wv] + lanePre;
    if (f && slot < CAP) {
      idxBuf[e * CAP + slot] = token;
      valBuf[e * CAP + slot] = w;
    }
    __syncthreads();
    if (tid == 0) sBase += sTot;
    __syncthreads();
  }
  int cnt = sBase;   // pad unused slots (expert underflow): token 0, weight 0
  for (int s = cnt + tid; s < CAP; s += 1024) {
    idxBuf[e * CAP + s] = 0;
    valBuf[e * CAP + s] = 0.f;
  }
}

// ---------------------------------------------------------------------------
// Kernel 3: fused expert FFN.  grid = (CAP/TM, E), 512 threads = 16 waves.
// TM=32 tokens/block halves per-block weight traffic vs TM=16; 16 waves keep
// per-wave accumulator footprint at 8 v8f tiles.  Wave w owns m-tile (w>>3)
// and an 8-tile column strip (w&7).
// WBF=true : weights pre-cast to bf16 in workspace (no cvt in hot loop).
// WBF=false: on-the-fly f32->bf16 conversion (fallback for small ws).
// k-outer / tile-inner loops: A fragment loaded once per k-step; independent
// WMMAs on distinct accumulators back-to-back.
// LDS: 32*(1024+8)*2 + 32*(256+8)*2 + small ~= 83 KB (<= 320 KB/WG on CDNA5).
// ---------------------------------------------------------------------------
template <bool WBF>
__global__ __launch_bounds__(NTHR) void moe_expert_kernel(
    const float* __restrict__ x,
    const float* __restrict__ W1f,  // [E, I, H] f32
    const float* __restrict__ W2f,  // [E, H, I] f32
    const bf16*  __restrict__ W1b,  // bf16 copies (valid iff WBF)
    const bf16*  __restrict__ W2b,
    const int*   __restrict__ idxBuf,
    const float* __restrict__ valBuf,
    float* __restrict__ out)
{
  __shared__ bf16  sTok[TM][Hdim + 8];      // +8 bf16 pad vs bank conflicts
  __shared__ bf16  sInter[TM][ICHUNK + 8];
  __shared__ int   sIdx[TM];
  __shared__ float sVal[TM];

  const int e    = blockIdx.y;
  const int t0   = blockIdx.x * TM;
  const int tid  = threadIdx.x;
  const int lane = tid & 31;
  const int wave = tid >> 5;            // 0..15
  const int mt   = wave >> 3;           // m-tile (0/1) owned by this wave
  const int iw   = wave & 7;            // column-strip index (0..7)
  const int row  = lane & 15;           // A/B row index and D column index
  const int k0b  = (lane >> 4) * 8;     // K sub-offset; also D row base (m)

  if (tid < TM) {
    sIdx[tid] = idxBuf[e * CAP + t0 + tid];
    sVal[tid] = valBuf[e * CAP + t0 + tid];
  }
  __syncthreads();

  // gather + f32->bf16 the 32 token rows (coalesced along H)
  for (int v = tid; v < TM * Hdim; v += NTHR) {
    int r = v >> 10;
    int c = v & (Hdim - 1);
    sTok[r][c] = (bf16)x[(size_t)sIdx[r] * Hdim + c];
  }
  __syncthreads();

  // output accumulators: 8 h-tiles per wave (8 waves cover H/16=64 per m-tile)
  v8f acc2[8];
#pragma unroll
  for (int j = 0; j < 8; ++j)
    acc2[j] = (v8f){0.f, 0.f, 0.f, 0.f, 0.f, 0.f, 0.f, 0.f};

  const size_t eW1 = (size_t)e * Idim * Hdim;
  const size_t eW2 = (size_t)e * Hdim * Idim;
  const bf16* aTok = sTok[mt * 16 + row];

  for (int ic = 0; ic < Idim; ic += ICHUNK) {
    // ---- phase 1: interChunk = relu(T @ W1[ic:ic+ICHUNK]^T) ----
    v8f acc1[2];
#pragma unroll
    for (int itl = 0; itl < 2; ++itl)
      acc1[itl] = (v8f){0.f, 0.f, 0.f, 0.f, 0.f, 0.f, 0.f, 0.f};

    for (int ks = 0; ks < Hdim; ks += 32) {
      v16bf a = frag_bf16(aTok, ks + k0b);
#pragma unroll
      for (int itl = 0; itl < 2; ++itl) {
        int it = iw * 2 + itl;                         // 16 i-tiles / 8 strips
        size_t roff = (size_t)(ic + it * 16 + row) * Hdim;
        v16bf b;
        if (WBF) b = frag_bf16(W1b + eW1 + roff, ks + k0b);
        else     b = frag_gf32(W1f + eW1 + roff, ks + k0b);
        acc1[itl] = __builtin_amdgcn_wmma_f32_16x16x32_bf16(
            false, a, false, b, (short)0, acc1[itl], false, false);
      }
    }
#pragma unroll
    for (int itl = 0; itl < 2; ++itl) {
      int it = iw * 2 + itl;
#pragma unroll
      for (int v = 0; v < 8; ++v) {                    // D: m=k0b+v, n=row
        float f = acc1[itl][v];
        f = f > 0.f ? f : 0.f;                         // ReLU
        sInter[mt * 16 + k0b + v][it * 16 + row] = (bf16)f;
      }
    }
    __syncthreads();

    // ---- phase 2: acc2 += interChunk @ W2[:, ic:ic+ICHUNK]^T ----
    const bf16* aInt = sInter[mt * 16 + row];
#pragma unroll
    for (int ks = 0; ks < ICHUNK; ks += 32) {
      v16bf a = frag_bf16(aInt, ks + k0b);
#pragma unroll
      for (int j = 0; j < 8; ++j) {
        int ht = iw * 8 + j;                           // 64 h-tiles / 8 strips
        size_t roff = (size_t)(ht * 16 + row) * Idim + ic;
        v16bf b;
        if (WBF) b = frag_bf16(W2b + eW2 + roff, ks + k0b);
        else     b = frag_gf32(W2f + eW2 + roff, ks + k0b);
        acc2[j] = __builtin_amdgcn_wmma_f32_16x16x32_bf16(
            false, a, false, b, (short)0, acc2[j], false, false);
      }
    }
    __syncthreads();                                   // before sInter reuse
  }

  // scale by router prob and scatter-add (<=2 adds per element -> determ.)
#pragma unroll
  for (int j = 0; j < 8; ++j) {
    int ht = iw * 8 + j;
#pragma unroll
    for (int v = 0; v < 8; ++v) {
      int m = mt * 16 + k0b + v;
      float val = sVal[m] * acc2[j][v];
      atomicAdd(out + (size_t)sIdx[m] * Hdim + ht * 16 + row, val);
    }
  }
}

// ---------------------------------------------------------------------------
extern "C" void kernel_launch(void* const* d_in, const int* in_sizes, int n_in,
                              void* d_out, int out_size, void* d_ws, size_t ws_size,
                              hipStream_t stream) {
  (void)in_sizes; (void)n_in; (void)out_size;
  const float* x   = (const float*)d_in[0];   // [B,S,H] f32
  const float* emb = (const float*)d_in[1];   // [E,H]
  const float* W1  = (const float*)d_in[2];   // [E,I,H]
  const float* W2  = (const float*)d_in[3];   // [E,H,I]
  float* out = (float*)d_out;                 // [B,S,H]

  // workspace layout
  const size_t wBytes   = (size_t)Ntok * Edim * 4;        // 1 MB   router probs
  const size_t idxBytes = (size_t)Edim * CAP * 4;         // 128 KB
  const size_t valBytes = (size_t)Edim * CAP * 4;         // 128 KB
  const size_t base     = wBytes + idxBytes + valBytes;   // 1.3 MB (256B aligned)
  const size_t wElems   = (size_t)Edim * Idim * Hdim;     // per weight tensor
  const size_t needBf   = base + 2 * wElems * sizeof(bf16); // ~68.4 MB

  char*  ws      = (char*)d_ws;
  float* weights = (float*)ws;
  int*   idxBuf  = (int*)(ws + wBytes);
  float* valBuf  = (float*)(ws + wBytes + idxBytes);
  bf16*  W1b     = (bf16*)(ws + base);
  bf16*  W2b     = (bf16*)(ws + base + wElems * sizeof(bf16));

  const bool useBf = (ws_size >= needBf);

  long n4 = (long)Ntok * Hdim / 4;
  moe_zero_kernel<<<dim3(2048), dim3(256), 0, stream>>>(out, n4);
  moe_router_kernel<<<dim3(Ntok / 8), dim3(256), 0, stream>>>(x, emb, weights);
  moe_assign_kernel<<<dim3(Edim), dim3(1024), 0, stream>>>(weights, idxBuf, valBuf);

  if (useBf) {
    moe_wcast_kernel<<<dim3(4096), dim3(256), 0, stream>>>(W1, W1b, (long)wElems);
    moe_wcast_kernel<<<dim3(4096), dim3(256), 0, stream>>>(W2, W2b, (long)wElems);
    moe_expert_kernel<true><<<dim3(CAP / TM, Edim), dim3(NTHR), 0, stream>>>(
        x, W1, W2, W1b, W2b, idxBuf, valBuf, out);
  } else {
    moe_expert_kernel<false><<<dim3(CAP / TM, Edim), dim3(NTHR), 0, stream>>>(
        x, W1, W2, W1b, W2b, idxBuf, valBuf, out);
  }
}
